// Model_15607911153665
// MI455X (gfx1250) — compile-verified, matrix operand order checked by reading, not run on hardware
//
#include <hip/hip_runtime.h>
#include <hip/hip_bf16.h>

// ---------------------------------------------------------------------------
// 2-layer GRU scan (B=2048, T=512, H=D=64) + FC head for MI455X (gfx1250).
//
// Roofline: state stream = 256MB (~11us @ 23.3TB/s); GEMM work = ~103 GFLOP
// -> f16 WMMA w/ f32 accum (v_wmma_f32_16x16x32_f16) keeps this tensor-bound.
// The recurrence serializes T=512 steps, so the critical path must not carry
// global-memory latency: x_t tiles are DOUBLE-BUFFERED into LDS with
// global_load_async_to_lds_b128 (ASYNCcnt) issued one step ahead, drained
// with s_wait_asynccnt + one barrier. Weights (f16) + fused biases live in
// LDS for the whole scan; h state lives in registers (WMMA C/D layout) and
// is re-fragmented via a 2KB per-wave LDS staging tile each step.
// 1 wave = 16 batch rows; 128-thread block = 64 rows -> 32 blocks.
// ---------------------------------------------------------------------------

typedef _Float16 half_t;
typedef __attribute__((ext_vector_type(16))) _Float16 v16h;
typedef __attribute__((ext_vector_type(8)))  _Float16 v8h;
typedef __attribute__((ext_vector_type(8)))  float    v8f;

#define HID 64
#define G3H 192              // 3 * HID
#define ROWS_PER_WAVE 16
#define ROWS_PER_BLK  64
#define NTHREADS      128

// --- dynamic LDS layout (bytes) --------------------------------------------
#define OFF_WIH1   0                        // 192*64 f16 = 24576
#define OFF_WHH1   (OFF_WIH1 + 24576)
#define OFF_WIH2   (OFF_WHH1 + 24576)
#define OFF_WHH2   (OFF_WIH2 + 24576)
#define OFF_BRZ1   (OFF_WHH2 + 24576)       // 128 f32
#define OFF_BIN1   (OFF_BRZ1 + 512)         // 64 f32
#define OFF_BHN1   (OFF_BIN1 + 256)         // 64 f32
#define OFF_BRZ2   (OFF_BHN1 + 256)
#define OFF_BIN2   (OFF_BRZ2 + 512)
#define OFF_BHN2   (OFF_BIN2 + 256)
#define OFF_XF32   (OFF_BHN2 + 256)         // 2 x (64*64 f32) = 32768 (dbl buf)
#define XBUF_BYTES 16384
#define OFF_HSTG   (OFF_XF32 + 2 * XBUF_BYTES) // 4 waves * (2 * 16*64 f16) = 16384
#define SMEM_BYTES (OFF_HSTG + 16384)       // = 149504

// Load a 16x32 f16 A/B fragment (v16h) from a row-major [rows][64] f16 array.
// 16-bit A layout (ISA 7.12.2): lane m (m=lane&15) holds row m; lanes 0-15
// carry K = {0..7, 16..23}, lanes 16-31 carry K = {8..15, 24..31} (+kbase).
// B (32x16) is symmetric with lane = column = weight row (since B = W^T).
__device__ __forceinline__ v16h load_frag(const half_t* base, int lane, int kbase) {
    const int row = lane & 15;
    const int g   = (lane >> 4) & 1;
    const half_t* p = base + row * HID + kbase + g * 8;
    v8h lo = *(const v8h*)(p);          // ds_load_b128
    v8h hi = *(const v8h*)(p + 16);     // ds_load_b128
    return __builtin_shufflevector(lo, hi, 0, 1, 2, 3, 4, 5, 6, 7,
                                           8, 9, 10, 11, 12, 13, 14, 15);
}

// Same fragment, but sourced from a row-major [16][64] f32 LDS tile
// (async-copied x_t), converting to f16 during assembly.
__device__ __forceinline__ v16h load_frag_f32(const float* base, int lane, int kbase) {
    const int row = lane & 15;
    const int g   = (lane >> 4) & 1;
    const float* p = base + row * HID + kbase + g * 8;
    v16h r;
#pragma unroll
    for (int i = 0; i < 8; ++i) {       // 4x ds_load_b128 + v_cvt_f16_f32
        r[i]     = (_Float16)p[i];
        r[i + 8] = (_Float16)p[16 + i];
    }
    return r;
}

__device__ __forceinline__ v8f wmma_f16(v16h a, v16h b, v8f c) {
    return __builtin_amdgcn_wmma_f32_16x16x32_f16(
        /*neg_a=*/false, a, /*neg_b=*/false, b,
        /*c_mod=*/(short)0, c, /*reuse_a=*/false, /*reuse_b=*/false);
}

__device__ __forceinline__ float sigmoid_f(float x) {
    return 1.0f / (1.0f + __expf(-x));
}
__device__ __forceinline__ float tanh_f(float x) {
    float e = __expf(-2.0f * x);
    return (1.0f - e) / (1.0f + e);
}

// One GRU cell step for 16 batch rows held by this wave (see round-0 notes).
__device__ __forceinline__ void gru_layer(
    v16h xa0, v16h xa1, half_t* hstage,
    const half_t* wih, const half_t* whh,
    const float* brz, const float* bin, const float* bhn,
    float (&hreg)[4][8], int lane)
{
    const int col   = lane & 15;
    const int rbase = ((lane >> 4) & 1) << 3;   // D-layout row base

    v16h ha0 = load_frag(hstage, lane, 0);
    v16h ha1 = load_frag(hstage, lane, 32);

    float rz[8][8];   // r tiles 0..3, z tiles 4..7 (cols 0..127 of the gates)
#pragma unroll
    for (int j = 0; j < 8; ++j) {
        float b = brz[j * 16 + col];
        v8f acc;
#pragma unroll
        for (int i = 0; i < 8; ++i) acc[i] = b;
        acc = wmma_f16(xa0, load_frag(wih + j * 16 * HID, lane, 0),  acc);
        acc = wmma_f16(xa1, load_frag(wih + j * 16 * HID, lane, 32), acc);
        acc = wmma_f16(ha0, load_frag(whh + j * 16 * HID, lane, 0),  acc);
        acc = wmma_f16(ha1, load_frag(whh + j * 16 * HID, lane, 32), acc);
#pragma unroll
        for (int i = 0; i < 8; ++i) rz[j][i] = sigmoid_f(acc[i]);
    }

#pragma unroll
    for (int j = 0; j < 4; ++j) {   // n-gate col tiles (weight rows 128..191)
        const half_t* wi = wih + (8 + j) * 16 * HID;
        const half_t* wh = whh + (8 + j) * 16 * HID;
        float bi = bin[j * 16 + col], bh = bhn[j * 16 + col];
        v8f ai, ah;
#pragma unroll
        for (int i = 0; i < 8; ++i) { ai[i] = bi; ah[i] = bh; }
        ai = wmma_f16(xa0, load_frag(wi, lane, 0),  ai);
        ai = wmma_f16(xa1, load_frag(wi, lane, 32), ai);
        ah = wmma_f16(ha0, load_frag(wh, lane, 0),  ah);
        ah = wmma_f16(ha1, load_frag(wh, lane, 32), ah);
#pragma unroll
        for (int i = 0; i < 8; ++i) {
            float n = tanh_f(ai[i] + rz[j][i] * ah[i]);
            float z = rz[4 + j][i];
            hreg[j][i] = (1.0f - z) * n + z * hreg[j][i];
        }
#pragma unroll
        for (int v = 0; v < 8; ++v)
            hstage[(v + rbase) * HID + j * 16 + col] = (half_t)hreg[j][v];
    }
}

__global__ __launch_bounds__(NTHREADS)
void gru2_wmma_kernel(const float* __restrict__ state,
                      const float* __restrict__ w_ih1, const float* __restrict__ w_hh1,
                      const float* __restrict__ b_ih1, const float* __restrict__ b_hh1,
                      const float* __restrict__ w_ih2, const float* __restrict__ w_hh2,
                      const float* __restrict__ b_ih2, const float* __restrict__ b_hh2,
                      const float* __restrict__ fc_w,  const float* __restrict__ fc_b,
                      float* __restrict__ out, int B, int T)
{
    extern __shared__ char smem[];
    const int tid  = threadIdx.x;
    const int wave = tid >> 5;
    const int lane = tid & 31;
    const int b0   = blockIdx.x * ROWS_PER_BLK;

    half_t* lwih1 = (half_t*)(smem + OFF_WIH1);
    half_t* lwhh1 = (half_t*)(smem + OFF_WHH1);
    half_t* lwih2 = (half_t*)(smem + OFF_WIH2);
    half_t* lwhh2 = (half_t*)(smem + OFF_WHH2);
    float*  brz1  = (float*)(smem + OFF_BRZ1);
    float*  bin1  = (float*)(smem + OFF_BIN1);
    float*  bhn1  = (float*)(smem + OFF_BHN1);
    float*  brz2  = (float*)(smem + OFF_BRZ2);
    float*  bin2  = (float*)(smem + OFF_BIN2);
    float*  bhn2  = (float*)(smem + OFF_BHN2);
    half_t* h1s   = (half_t*)(smem + OFF_HSTG) + wave * 2048;
    half_t* h2s   = h1s + 1024;

    // Generic-pointer low 32 bits = wave-relative LDS byte offset (flat
    // aperture); used as the async-copy dest.
    const unsigned smem_base = (unsigned)(uintptr_t)(void*)smem;
    const unsigned long long state_base = (unsigned long long)(uintptr_t)state;

    // Issue one step's x tile (64 rows x 64 f32) as 8 async B128 copies per
    // thread: thread -> (row = tid/2, cols = (tid&1)*32 .. +31), 128B contig.
    auto issue_x = [&](int t, int buf) {
        const int r  = tid >> 1;
        const int cb = (tid & 1) * 32;
        const unsigned voff = (unsigned)((((b0 + r) * T + t) * HID + cb) * 4);
        const unsigned ldsb = smem_base + OFF_XF32 + (unsigned)buf * XBUF_BYTES
                            + (unsigned)(r * HID + cb) * 4u;
        asm volatile(
            "global_load_async_to_lds_b128 %0, %1, %2 offset:0\n\t"
            "global_load_async_to_lds_b128 %0, %1, %2 offset:16\n\t"
            "global_load_async_to_lds_b128 %0, %1, %2 offset:32\n\t"
            "global_load_async_to_lds_b128 %0, %1, %2 offset:48\n\t"
            "global_load_async_to_lds_b128 %0, %1, %2 offset:64\n\t"
            "global_load_async_to_lds_b128 %0, %1, %2 offset:80\n\t"
            "global_load_async_to_lds_b128 %0, %1, %2 offset:96\n\t"
            "global_load_async_to_lds_b128 %0, %1, %2 offset:112"
            :: "v"(ldsb), "v"(voff), "s"(state_base) : "memory");
    };

    // Kick off t=0 copy before anything else so it overlaps weight staging.
    issue_x(0, 0);

    // --- stage weights (f32 -> f16) and fused biases into LDS --------------
    {
        const float* gw[4] = { w_ih1, w_hh1, w_ih2, w_hh2 };
        half_t*      lw[4] = { lwih1, lwhh1, lwih2, lwhh2 };
#pragma unroll
        for (int m = 0; m < 4; ++m)
            for (int i = tid; i < G3H * HID; i += NTHREADS)
                lw[m][i] = (half_t)gw[m][i];
        if (tid < 128) { brz1[tid] = b_ih1[tid] + b_hh1[tid];
                         brz2[tid] = b_ih2[tid] + b_hh2[tid]; }
        if (tid < 64)  { bin1[tid] = b_ih1[128 + tid]; bhn1[tid] = b_hh1[128 + tid];
                         bin2[tid] = b_ih2[128 + tid]; bhn2[tid] = b_hh2[128 + tid]; }
    }

    // --- init hidden state (regs + staging) to zero -------------------------
    float h1reg[4][8] = {{0}}, h2reg[4][8] = {{0}};
    for (int i = lane; i < 1024; i += 32) { h1s[i] = (half_t)0.0f; h2s[i] = (half_t)0.0f; }

    // --- scan over T (software-pipelined x prefetch) ------------------------
#pragma unroll 1
    for (int t = 0; t < T; ++t) {
        // Drain this wave's async copies for buffer t&1, then make them
        // visible block-wide. (Reads of the *other* buffer from step t-1 were
        // consumed before this barrier, so re-issuing into it below is safe.)
        asm volatile("s_wait_asynccnt 0x0" ::: "memory");
        __syncthreads();
        if (t + 1 < T) issue_x(t + 1, (t + 1) & 1);

        const float* xrows = (const float*)(smem + OFF_XF32)
                           + (t & 1) * (XBUF_BYTES / 4)
                           + wave * ROWS_PER_WAVE * HID;
        v16h xa0 = load_frag_f32(xrows, lane, 0);
        v16h xa1 = load_frag_f32(xrows, lane, 32);

        gru_layer(xa0, xa1, h1s, lwih1, lwhh1, brz1, bin1, bhn1, h1reg, lane);

        // layer-2 input = new h1 (just written to staging by this wave)
        v16h s1a0 = load_frag(h1s, lane, 0);
        v16h s1a1 = load_frag(h1s, lane, 32);

        gru_layer(s1a0, s1a1, h2s, lwih2, lwhh2, brz2, bin2, bhn2, h2reg, lane);
    }

    // --- FC head: out = h2 @ fc_w^T + fc_b  ([64,64]x[64,8] per block) ------
    __syncthreads();                       // done with weight LDS; reuse it
    float* h2f = (float*)smem + wave * 1024;   // row r lives at [r*64 ..]
    {
        const int col   = lane & 15;
        const int rbase = ((lane >> 4) & 1) << 3;
#pragma unroll
        for (int j = 0; j < 4; ++j)
#pragma unroll
            for (int v = 0; v < 8; ++v)
                h2f[(v + rbase) * HID + j * 16 + col] = h2reg[j][v];
    }
    __syncthreads();
    for (int o = tid; o < ROWS_PER_BLK * 8; o += NTHREADS) {
        int r = o >> 3, a = o & 7;
        const float* hrow = (float*)smem + r * HID;
        float s = fc_b[a];
#pragma unroll 8
        for (int k = 0; k < HID; ++k) s += hrow[k] * fc_w[a * HID + k];
        out[(size_t)(b0 + r) * 8 + a] = s;
    }
}

extern "C" void kernel_launch(void* const* d_in, const int* in_sizes, int n_in,
                              void* d_out, int out_size, void* d_ws, size_t ws_size,
                              hipStream_t stream) {
    const float* state = (const float*)d_in[0];
    const float* w_ih1 = (const float*)d_in[1];
    const float* w_hh1 = (const float*)d_in[2];
    const float* b_ih1 = (const float*)d_in[3];
    const float* b_hh1 = (const float*)d_in[4];
    const float* w_ih2 = (const float*)d_in[5];
    const float* w_hh2 = (const float*)d_in[6];
    const float* b_ih2 = (const float*)d_in[7];
    const float* b_hh2 = (const float*)d_in[8];
    const float* fc_w  = (const float*)d_in[9];
    const float* fc_b  = (const float*)d_in[10];
    float* out = (float*)d_out;

    const int B = out_size / 8;                   // 2048
    const int T = in_sizes[0] / (B * HID);        // 512

    dim3 grid(B / ROWS_PER_BLK), block(NTHREADS);
    gru2_wmma_kernel<<<grid, block, SMEM_BYTES, stream>>>(
        state, w_ih1, w_hh1, b_ih1, b_hh1,
        w_ih2, w_hh2, b_ih2, b_hh2, fc_w, fc_b, out, B, T);
}
